// BaselineRNN_23227183137368
// MI455X (gfx1250) — matrix operand order repaired and checked by reading, not compiled, not run on hardware
//
#include <hip/hip_runtime.h>
#include <hip/hip_bf16.h>

typedef __attribute__((ext_vector_type(16))) _Float16 v16h;
typedef __attribute__((ext_vector_type(8)))  float    v8f;

#define T_STEPS  512
#define NLAYERS  8
#define HID      24
#define WPB      4          // waves per block (128 threads)
#define NWAVES   256        // 4096 batch / 16 per wave
// ds_swizzle offset: group-of-32, xor_mask=0x10, or=0, and=0x1f  -> SWAPX16
#define SWZ_XOR16 0x401F

union TileU { uint4 q[2]; unsigned int u[8]; v16h h; };

__device__ __forceinline__ float swz_xor16(float x) {
    return __int_as_float(__builtin_amdgcn_ds_swizzle(__float_as_int(x), SWZ_XOR16));
}

// Branchless tanh: hardware V_TANH_F32 if the builtin exists, else
// tanh(x) = 1 - 2/(exp2(2*log2e*x)+1) with raw v_exp_f32 / v_rcp_f32.
__device__ __forceinline__ float fast_tanh(float x) {
#if __has_builtin(__builtin_amdgcn_tanhf)
    return __builtin_amdgcn_tanhf(x);
#else
    const float e = __builtin_amdgcn_exp2f(x * 2.8853900817779268f); // 2*log2(e)
    return 1.0f - 2.0f * __builtin_amdgcn_rcpf(e + 1.0f);
#endif
}

__device__ __forceinline__ v16h zero_v16h() {
    v16h z;
#pragma unroll
    for (int i = 0; i < 16; ++i) z[i] = (_Float16)0.0f;
    return z;
}

__device__ __forceinline__ v8f zero_v8f() {
    v8f z;
#pragma unroll
    for (int i = 0; i < 8; ++i) z[i] = 0.0f;
    return z;
}

__global__ __launch_bounds__(128, 1)
void rnn_fused_wmma(const float* __restrict__ x,      // (4096, 512, 6)
                    const float* __restrict__ Wih0,   // (24, 6)
                    const float* __restrict__ Wihr,   // (7, 24, 24)
                    const float* __restrict__ Whh,    // (8, 24, 24)
                    const float* __restrict__ bih,    // (8, 24)
                    const float* __restrict__ bhh,    // (8, 24)
                    const float* __restrict__ fcw,    // (3, 24)
                    const float* __restrict__ fcb,    // (3,)
                    float* __restrict__ out)          // (4096, 3)
{
    // Prepacked A-matrix tiles (f16, exact WMMA A layout), 4 tiles/layer:
    // kind 0/1 = Wih(+bias col) M-tiles 0..15 / 16..23, kind 2/3 = Whh M-tiles.
    __shared__ __align__(16) unsigned int lds_w[NLAYERS * 4 * 32 * 8];   // 32 KB
    __shared__ float lds_h[WPB][16][HID + 1];                            // FC staging

    const int tid = threadIdx.x;

    // ---------------- prepack weights into LDS (A-matrix layout) -------------
    // A order used here (16-bit, 16x32, contiguous K, B matches): lane L (half=L>>4,row=L&15) holds row M=row of
// tile; VGPR v element pair covers K = 2v + 16*half + {0,1}.
    for (int idx = tid; idx < NLAYERS * 4 * 32 * 8; idx += blockDim.x) {
        const int v    = idx & 7;
        const int ln   = (idx >> 3) & 31;
        const int kind = (idx >> 8) & 3;
        const int l    = idx >> 10;
        const int half = ln >> 4, row = ln & 15;
        const int m    = (kind & 1) * 16 + row;
        const int is_hh = kind >> 1;
        unsigned int packed = 0;
#pragma unroll
        for (int j = 0; j < 2; ++j) {
            const int K = 2 * v + 16 * half + j;
            float val = 0.0f;
            if (m < HID) {
                if (is_hh) {
                    if (K < HID) val = Whh[(l * HID + m) * HID + K];
                } else {
                    const int inK = (l == 0) ? 6 : HID;
                    if (K < inK)
                        val = (l == 0) ? Wih0[m * 6 + K]
                                       : Wihr[((l - 1) * HID + m) * HID + K];
                    else if (K == inK)   // folded bias column (B supplies 1.0)
                        val = bih[l * HID + m] + bhh[l * HID + m];
                }
            }
            _Float16 hv = (_Float16)val;
            packed |= (unsigned int)__builtin_bit_cast(unsigned short, hv) << (16 * j);
        }
        lds_w[idx] = packed;
    }
    __syncthreads();

    // ---------------- fused 8-layer RNN scan ---------------------------------
    const int lane  = tid & 31;
    const int wib   = tid >> 5;
    const int wgid  = blockIdx.x * WPB + wib;       // 0..255
    const int bbase = wgid * 16;                    // 16 batch rows per wave
    const int brow  = lane & 15;
    const bool lo   = lane < 16;
    const size_t xrow = (size_t)(bbase + brow) * T_STEPS * 6;

    const v8f c_zero = zero_v8f();

    // Hoist all weight A-tiles into registers (loop-invariant over t).
    // 32 tiles x 8 VGPRs = 256 VGPRs; CDNA5 allows up to 1024 VGPRs/wave.
    TileU wih0t[NLAYERS], wih1t[NLAYERS], whh0t[NLAYERS], whh1t[NLAYERS];
#pragma unroll
    for (int l = 0; l < NLAYERS; ++l) {
        const uint4* p0 = (const uint4*)&lds_w[((l * 4 + 0) * 32 + lane) * 8];
        const uint4* p1 = (const uint4*)&lds_w[((l * 4 + 1) * 32 + lane) * 8];
        const uint4* p2 = (const uint4*)&lds_w[((l * 4 + 2) * 32 + lane) * 8];
        const uint4* p3 = (const uint4*)&lds_w[((l * 4 + 3) * 32 + lane) * 8];
        wih0t[l].q[0] = p0[0]; wih0t[l].q[1] = p0[1];
        wih1t[l].q[0] = p1[0]; wih1t[l].q[1] = p1[1];
        whh0t[l].q[0] = p2[0]; whh0t[l].q[1] = p2[1];
        whh1t[l].q[0] = p3[0]; whh1t[l].q[1] = p3[1];
    }

    // Recurrent hidden states, kept in WMMA B-operand layout (f16).
    v16h hstate[NLAYERS];
#pragma unroll
    for (int l = 0; l < NLAYERS; ++l) hstate[l] = zero_v16h();

    // Software-pipelined x load: keep the load for step t+1 in flight across
    // the whole 8-layer WMMA chain of step t.
    float xv[6];
#pragma unroll
    for (int i = 0; i < 6; ++i) xv[i] = x[xrow + i];

#pragma unroll 1
    for (int t = 0; t < T_STEPS; ++t) {
        // Issue next step's load first (uniform clamped index, no divergence).
        const int tn = (t + 1 < T_STEPS) ? (t + 1) : t;
        const float* xpn = x + xrow + (size_t)tn * 6;
        float xn[6];
#pragma unroll
        for (int i = 0; i < 6; ++i) xn[i] = xpn[i];

        // ---- build x_t B-operand: K=0..5 = x, K=6 = 1.0 (bias), rest 0 -----
        v16h bprev = zero_v16h();
#pragma unroll
        for (int i = 0; i < 6; ++i) bprev[i] = (_Float16)(lo ? xv[i] : 0.0f);
        bprev[6] = (_Float16)(lo ? 1.0f : 0.0f);

#pragma unroll
        for (int l = 0; l < NLAYERS; ++l) {
            // h_new^T = Wih*in^T (+bias via 1.0 column) + Whh*h^T
            v8f c0 = __builtin_amdgcn_wmma_f32_16x16x32_f16(
                false, wih0t[l].h, false, bprev, (short)0, c_zero, false, false);
            c0 = __builtin_amdgcn_wmma_f32_16x16x32_f16(
                false, whh0t[l].h, false, hstate[l], (short)0, c0, false, false);
            v8f c1 = __builtin_amdgcn_wmma_f32_16x16x32_f16(
                false, wih1t[l].h, false, bprev, (short)0, c_zero, false, false);
            c1 = __builtin_amdgcn_wmma_f32_16x16x32_f16(
                false, whh1t[l].h, false, hstate[l], (short)0, c1, false, false);

            // tanh activation on both M-tiles (single-instruction trans ops).
            float t0[8], t1[8];
#pragma unroll
            for (int r = 0; r < 8; ++r) { t0[r] = fast_tanh(c0[r]); t1[r] = fast_tanh(c1[r]); }

            // Convert C/D layout -> B-operand layout with one xor-16 lane swap:
            //  lane<16 needs hidden 8..15 (held by lane+16 in tile0)
            //  lane>=16 needs hidden 16..23 (held by lane-16 in tile1)
            float sh[8];
#pragma unroll
            for (int r = 0; r < 8; ++r) sh[r] = swz_xor16(lo ? t1[r] : t0[r]);

            v16h bn;
#pragma unroll
            for (int e = 0; e < 8; ++e)                     // K = e  | 16+e
                bn[e] = (_Float16)(lo ? t0[e] : sh[e]);
#pragma unroll
            for (int e = 0; e < 8; ++e)                     // K = 8+e | 24+e
                bn[8 + e] = (_Float16)(lo ? sh[e] : ((e == 0) ? 1.0f : 0.0f));
            // bn[8] on upper lanes is the K=24 bias-1.0 column for the next
            // layer's Wih GEMM; Whh columns >=24 are zero so it is inert there.

            hstate[l] = bn;
            bprev = bn;
        }

        // Rotate the prefetched x into place (wait lands after full step).
#pragma unroll
        for (int i = 0; i < 6; ++i) xv[i] = xn[i];
    }

    // ---------------- FC head on the final hidden state ----------------------
#pragma unroll
    for (int e = 0; e < 16; ++e)
        if (lo) lds_h[wib][brow][e] = (float)hstate[NLAYERS - 1][e];
#pragma unroll
    for (int e = 0; e < 8; ++e)
        if (!lo) lds_h[wib][brow][16 + e] = (float)hstate[NLAYERS - 1][e];
    __syncthreads();

    if (lo) {
        float a0 = fcb[0], a1 = fcb[1], a2 = fcb[2];
#pragma unroll
        for (int h = 0; h < HID; ++h) {
            const float hv = lds_h[wib][brow][h];
            a0 += hv * fcw[h];
            a1 += hv * fcw[HID + h];
            a2 += hv * fcw[2 * HID + h];
        }
        float* op = out + (size_t)(bbase + brow) * 3;
        op[0] = a0; op[1] = a1; op[2] = a2;
    }
}

extern "C" void kernel_launch(void* const* d_in, const int* in_sizes, int n_in,
                              void* d_out, int out_size, void* d_ws, size_t ws_size,
                              hipStream_t stream) {
    (void)in_sizes; (void)n_in; (void)d_ws; (void)ws_size; (void)out_size;
    const float* xx   = (const float*)d_in[0];
    const float* Wih0 = (const float*)d_in[1];
    const float* Wihr = (const float*)d_in[2];
    const float* Whh  = (const float*)d_in[3];
    const float* bih  = (const float*)d_in[4];
    const float* bhh  = (const float*)d_in[5];
    const float* fcw  = (const float*)d_in[6];
    const float* fcb  = (const float*)d_in[7];
    float* out = (float*)d_out;

    dim3 grid(NWAVES / WPB);   // 64 blocks
    dim3 block(WPB * 32);      // 128 threads = 4 wave32
    rnn_fused_wmma<<<grid, block, 0, stream>>>(xx, Wih0, Wihr, Whh, bih, bhh,
                                               fcw, fcb, out);
}
